// DDSPVCProxy_73091753443823
// MI455X (gfx1250) — matrix operand order repaired and checked by reading, not compiled, not run on hardware
//
#include <hip/hip_runtime.h>
#include <hip/hip_bf16.h>
#include <math.h>

// ---------------- constants matching the reference ----------------
#define SR      16000
#define HOP     256
#define NFFT    2048
#define NMELS   80
#define NSTFT   1025          // NFFT/2+1
#define MIN_P   32            // SR/500
#define FRAME_LEN 512         // 2*HOP
#define NB      16            // batch
#define TLEN    131072        // samples
#define NF      513           // frames (both pitch framing and STFT framing)
#define NROWS   (NB*NF)       // 8208
#define LOLA    (NFFT + HOP*(NF-1))   // 133120 overlap-add length
#define GL_MOM  (0.99f/1.99f)

typedef __attribute__((ext_vector_type(16))) _Float16 v16h;
typedef __attribute__((ext_vector_type(8)))  _Float16 v8h;
typedef __attribute__((ext_vector_type(8)))  float    v8f;

union AFrag { v16h v; v8h h[2]; };

__device__ __forceinline__ float hannw(int i) {
    return 0.5f - 0.5f * __cosf(6.2831853071795864f * (float)i / (float)NFFT);
}
__device__ __forceinline__ float eluf(float v) {
    return v > 0.0f ? v : (__expf(v) - 1.0f);
}
__device__ __forceinline__ float u01hash(unsigned x) {
    x ^= x >> 17; x *= 0xed5ad4bbu; x ^= x >> 11; x *= 0xac4c1b51u;
    x ^= x >> 15; x *= 0x31848babu; x ^= x >> 14;
    return (float)(x >> 8) * (1.0f / 16777216.0f);
}

// ---------------- 2048-pt complex FFT in LDS (256 threads) ----------------
__device__ void fft2048(float2* S, float sign) {
    #pragma unroll 1
    for (int st = 1; st <= 11; ++st) {
        const int half = 1 << (st - 1);
        const int m = half << 1;
        for (int bf = threadIdx.x; bf < 1024; bf += 256) {
            int j   = bf & (half - 1);
            int grp = bf >> (st - 1);
            int i0  = grp * m + j;
            int i1  = i0 + half;
            float ang = sign * 3.14159265358979f * (float)j / (float)half;
            float sn, cs; __sincosf(ang, &sn, &cs);
            float2 u = S[i0], v = S[i1];
            float tr = v.x * cs - v.y * sn;
            float ti = v.x * sn + v.y * cs;
            S[i0] = make_float2(u.x + tr, u.y + ti);
            S[i1] = make_float2(u.x - tr, u.y - ti);
        }
        __syncthreads();
    }
}

// ---------------- pitch / loudness features ----------------
__global__ __launch_bounds__(256) void pitch_kernel(const float* __restrict__ x,
                                                    float* __restrict__ feats) {
    __shared__ float fr[FRAME_LEN];
    __shared__ float red[256];
    __shared__ int   redi[256];
    const int fidx = blockIdx.x;
    const int b = fidx / NF, t = fidx - b * NF;
    const float* xb = x + (long)b * TLEN;
    const int base = t * HOP - FRAME_LEN / 2;
    for (int i = threadIdx.x; i < FRAME_LEN; i += 256) {
        int j = base + i;
        if (j < 0) j = -j;
        if (j >= TLEN) j = 2 * TLEN - 2 - j;
        fr[i] = xb[j];
    }
    __syncthreads();
    float ss = 0.0f;
    for (int i = threadIdx.x; i < FRAME_LEN; i += 256) ss += fr[i] * fr[i];
    red[threadIdx.x] = ss;
    __syncthreads();
    for (int s = 128; s > 0; s >>= 1) {
        if (threadIdx.x < s) red[threadIdx.x] += red[threadIdx.x + s];
        __syncthreads();
    }
    const float sumsq = red[0];
    const float inv = 1.0f / (sqrtf(sumsq) + 1e-8f);
    __syncthreads();
    // autocorrelation of normalized frame, lags [32, 256)
    float best = -1e30f; int ibest = 0;
    if (threadIdx.x < 224) {
        const int lag = MIN_P + threadIdx.x;
        float s = 0.0f;
        for (int n = 0; n + lag < FRAME_LEN; ++n) s += fr[n] * fr[n + lag];
        best = s * inv * inv;
        ibest = threadIdx.x;
    }
    red[threadIdx.x] = best; redi[threadIdx.x] = ibest;
    __syncthreads();
    for (int s = 128; s > 0; s >>= 1) {
        if (threadIdx.x < s) {
            float a = red[threadIdx.x], bb = red[threadIdx.x + s];
            int ia = redi[threadIdx.x], ib = redi[threadIdx.x + s];
            if (bb > a || (bb == a && ib < ia)) { red[threadIdx.x] = bb; redi[threadIdx.x] = ib; }
        }
        __syncthreads();
    }
    if (threadIdx.x == 0) {
        float period = (float)(redi[0] + MIN_P);
        float f0 = (float)SR / (period + 1e-8f);
        f0 = fminf(fmaxf(f0, 50.0f), 500.0f);
        float loud = 20.0f * log10f(sqrtf(sumsq / (float)FRAME_LEN) + 1e-8f);
        feats[fidx * 2 + 0] = f0;
        feats[fidx * 2 + 1] = loud;
    }
}

// ---------------- MLP layer 1 (2->256) ----------------
__global__ void h1_kernel(const float* __restrict__ feats, const float* __restrict__ W1,
                          const float* __restrict__ b1, _Float16* __restrict__ h1h) {
    int gid = blockIdx.x * blockDim.x + threadIdx.x;
    if (gid >= NROWS * 256) return;
    int r = gid >> 8, j = gid & 255;
    float v = feats[r * 2] * W1[j] + feats[r * 2 + 1] * W1[256 + j] + b1[j];
    h1h[gid] = (_Float16)eluf(v);
}

// Pack a row-major f32 [K x N] weight matrix into per-lane WMMA B-fragment order:
// dst[(( (k>>5)*2 + ((k>>3)&1) )*N + n)*16 + ((k&7)|((k>>1)&8))] -- so each lane's fragment
// (16 K values in WMMA table order at one column) is one contiguous, 32B-aligned vector.
__global__ void packB_kernel(const float* __restrict__ src, _Float16* __restrict__ dst,
                             int K, int N) {
    int gid = blockIdx.x * blockDim.x + threadIdx.x;
    if (gid >= K * N) return;
    int k = gid / N, n = gid - k * N;
    int grp = (k >> 5) * 2 + ((k >> 3) & 1);
    dst[((long)grp * N + n) * 16 + ((k & 7) | ((k >> 1) & 8))] = (_Float16)src[gid];
}

// ---------------- WMMA GEMM: h2 = elu(h1[8208x256] @ W2[256x256] + b2) ----------------
// One wave computes one 16x16 output tile; wave w handles N-tiles {w, w+8}.
__global__ __launch_bounds__(256) void gemm2_kernel(const _Float16* __restrict__ A,
                                                    const _Float16* __restrict__ Bp,
                                                    const float* __restrict__ bias,
                                                    _Float16* __restrict__ Out) {
    const int lane = threadIdx.x & 31;
    const int wv   = threadIdx.x >> 5;
    const int m    = lane & 15;
    const int hi   = lane >> 4;      // half-wave select per ISA layout
    const int r0   = blockIdx.x * 16;
    const _Float16* Arow = A + (long)(r0 + m) * 256;
    #pragma unroll 1
    for (int ntp = 0; ntp < 2; ++ntp) {
        const int n0 = (wv + ntp * 8) * 16;
        const int n  = n0 + m;
        v8f acc = {};
        #pragma unroll 1
        for (int kk = 0; kk < 256; kk += 32) {
            __builtin_prefetch(Arow + kk + 32, 0, 3);
            AFrag a;
            a.h[0] = *(const v8h*)(Arow + kk + hi * 8);        // K = kk+8hi .. +7
            a.h[1] = *(const v8h*)(Arow + kk + 16 + hi * 8);   // K = kk+16+8hi .. +7
            const int grp = (kk >> 5) * 2 + hi;
            v16h b = *(const v16h*)(Bp + ((long)grp * 256 + n) * 16);
            acc = __builtin_amdgcn_wmma_f32_16x16x32_f16(false, a.v, false, b,
                                                         (short)0, acc, false, false);
        }
        #pragma unroll
        for (int r = 0; r < 8; ++r) {
            int row = r0 + r + hi * 8;
            float v = acc[r] + bias[n];
            Out[(long)row * 256 + n] = (_Float16)eluf(v);
        }
    }
}

// ---------------- WMMA GEMM: content = h2[8208x256] @ W3[256x64] + b3 ----------------
__global__ __launch_bounds__(128) void gemm3_kernel(const _Float16* __restrict__ A,
                                                    const _Float16* __restrict__ Bp,
                                                    const float* __restrict__ bias,
                                                    float* __restrict__ Out) {
    const int lane = threadIdx.x & 31;
    const int wv   = threadIdx.x >> 5;   // 0..3 -> N tiles
    const int m    = lane & 15;
    const int hi   = lane >> 4;
    const int r0   = blockIdx.x * 16;
    const int n    = wv * 16 + m;
    const _Float16* Arow = A + (long)(r0 + m) * 256;
    v8f acc = {};
    #pragma unroll 1
    for (int kk = 0; kk < 256; kk += 32) {
        AFrag a;
        a.h[0] = *(const v8h*)(Arow + kk + hi * 8);
        a.h[1] = *(const v8h*)(Arow + kk + 16 + hi * 8);
        const int grp = (kk >> 5) * 2 + hi;
        v16h b = *(const v16h*)(Bp + ((long)grp * 64 + n) * 16);
        acc = __builtin_amdgcn_wmma_f32_16x16x32_f16(false, a.v, false, b,
                                                     (short)0, acc, false, false);
    }
    #pragma unroll
    for (int r = 0; r < 8; ++r) {
        int row = r0 + r + hi * 8;
        Out[(long)row * 64 + n] = acc[r] + bias[n];
    }
}

// ---------------- forward STFT (mode 0: magnitude; mode 1: Griffin-Lim update) ----------------
__global__ __launch_bounds__(256) void stft_kernel(const float* __restrict__ sig, int mode,
                                                   float* __restrict__ mag,
                                                   float2* __restrict__ angles,
                                                   float2* __restrict__ tprev) {
    __shared__ float2 S[NFFT];
    const int fidx = blockIdx.x;
    const int b = fidx / NF, t = fidx - b * NF;
    const float* xb = sig + (long)b * TLEN;
    const int base = t * HOP - NFFT / 2;
    for (int i = threadIdx.x; i < NFFT; i += 256) {
        int j = base + i;
        if (j < 0) j = -j;
        if (j >= TLEN) j = 2 * TLEN - 2 - j;
        float val = xb[j] * hannw(i);
        int rp = __brev((unsigned)i) >> 21;     // 11-bit reverse
        S[rp] = make_float2(val, 0.0f);
    }
    __syncthreads();
    fft2048(S, -1.0f);
    if (mode == 0) {
        for (int k = threadIdx.x; k < NSTFT; k += 256) {
            float2 X = S[k];
            mag[(long)fidx * NSTFT + k] = sqrtf(X.x * X.x + X.y * X.y);
        }
    } else {
        for (int k = threadIdx.x; k < NSTFT; k += 256) {
            long idx = (long)fidx * NSTFT + k;
            float2 X = S[k];
            float2 tp = tprev[idx];
            float ar = X.x - GL_MOM * tp.x;
            float ai = X.y - GL_MOM * tp.y;
            float rn = 1.0f / (sqrtf(ar * ar + ai * ai) + 1e-16f);
            angles[idx] = make_float2(ar * rn, ai * rn);
            tprev[idx]  = X;
        }
    }
}

// ---------------- inverse STFT per-frame (spec*angles -> windowed time frames) ----------------
__global__ __launch_bounds__(256) void istft_kernel(const float* __restrict__ lin,
                                                    const float2* __restrict__ angles,
                                                    float* __restrict__ framesOut) {
    __shared__ float2 S[NFFT];
    const int fidx = blockIdx.x;
    for (int i = threadIdx.x; i < NFFT; i += 256) {
        int k = (i <= NFFT / 2) ? i : NFFT - i;
        long idx = (long)fidx * NSTFT + k;
        float mg = lin[idx];
        float2 a = angles[idx];
        float re = mg * a.x, im = mg * a.y;
        if (i > NFFT / 2) im = -im;             // Hermitian extension
        int rp = __brev((unsigned)i) >> 21;
        S[rp] = make_float2(re, im);
    }
    __syncthreads();
    fft2048(S, +1.0f);
    const float scale = 1.0f / (float)NFFT;
    for (int i = threadIdx.x; i < NFFT; i += 256) {
        framesOut[(long)fidx * NFFT + i] = S[i].x * scale * hannw(i);
    }
}

// ---------------- overlap-add envelope (window^2 sum) ----------------
__global__ void env_kernel(float* __restrict__ env) {
    int l = blockIdx.x * blockDim.x + threadIdx.x;
    if (l >= LOLA) return;
    int fmax = l >> 8, fmin = fmax - 7;
    if (fmin < 0) fmin = 0;
    if (fmax > NF - 1) fmax = NF - 1;
    float acc = 0.0f;
    for (int f = fmin; f <= fmax; ++f) {
        float w = hannw(l - f * HOP);
        acc += w * w;
    }
    env[l] = acc > 1e-11f ? acc : 1.0f;
}

// ---------------- overlap-add gather + trim ----------------
__global__ void ola_kernel(const float* __restrict__ frames, const float* __restrict__ env,
                           float* __restrict__ out) {
    long gid = (long)blockIdx.x * blockDim.x + threadIdx.x;
    if (gid >= (long)NB * TLEN) return;
    int b = (int)(gid / TLEN);
    int t = (int)(gid - (long)b * TLEN);
    int l = t + NFFT / 2;
    int fmax = l >> 8, fmin = fmax - 7;
    if (fmin < 0) fmin = 0;
    if (fmax > NF - 1) fmax = NF - 1;
    float acc = 0.0f;
    for (int f = fmin; f <= fmax; ++f)
        acc += frames[((long)b * NF + f) * NFFT + (l - f * HOP)];
    out[gid] = acc / env[l];
}

// ---------------- mel filterbank ----------------
__device__ __forceinline__ float mel_fpt(int p, float mmax) {
    return 700.0f * (powf(10.0f, (float)p * mmax / (81.0f * 2595.0f)) - 1.0f);
}
__global__ void fb_kernel(float* __restrict__ fb) {
    int gid = blockIdx.x * blockDim.x + threadIdx.x;
    if (gid >= NSTFT * NMELS) return;
    int f = gid / NMELS, m = gid - f * NMELS;
    float freq = (float)f * ((float)SR * 0.5f / (float)(NSTFT - 1));
    float mmax = 2595.0f * log10f(1.0f + (float)SR * 0.5f / 700.0f);
    float p0 = mel_fpt(m, mmax), p1 = mel_fpt(m + 1, mmax), p2 = mel_fpt(m + 2, mmax);
    float down = (freq - p0) / (p1 - p0);
    float up   = (p2 - freq) / (p2 - p1);
    fb[gid] = fmaxf(0.0f, fminf(down, up));
}

__global__ void mel_kernel(const float* __restrict__ mag, const float* __restrict__ fb,
                           float* __restrict__ mel) {
    int gid = blockIdx.x * blockDim.x + threadIdx.x;
    if (gid >= NROWS * NMELS) return;
    int fidx = gid / NMELS, m = gid - fidx * NMELS;
    const float* mg = mag + (long)fidx * NSTFT;
    float acc = 0.0f;
    for (int f = 0; f < NSTFT; ++f) acc += mg[f] * fb[f * NMELS + m];
    mel[gid] = acc;
}

__global__ void m2l_kernel(const float* __restrict__ mel, float* __restrict__ lin) {
    long gid = (long)blockIdx.x * blockDim.x + threadIdx.x;
    if (gid >= (long)NROWS * NSTFT) return;
    int fidx = (int)(gid / NSTFT);
    int f = (int)(gid - (long)fidx * NSTFT);
    float src = fmaxf(((float)f + 0.5f) * ((float)NMELS / (float)NSTFT) - 0.5f, 0.0f);
    int i0 = (int)floorf(src);
    if (i0 > NMELS - 1) i0 = NMELS - 1;
    int i1 = i0 + 1; if (i1 > NMELS - 1) i1 = NMELS - 1;
    float w = src - (float)i0;
    lin[gid] = (1.0f - w) * mel[(long)fidx * NMELS + i0] + w * mel[(long)fidx * NMELS + i1];
}

// ---------------- Griffin-Lim phase init ----------------
__global__ void ainit_kernel(float2* __restrict__ angles, float2* __restrict__ tprev) {
    long gid = (long)blockIdx.x * blockDim.x + threadIdx.x;
    if (gid >= (long)NROWS * NSTFT) return;
    unsigned u = (unsigned)gid;
    angles[gid] = make_float2(u01hash(u * 2u + 1u), u01hash(u * 2u + 0x9e3779b9u));
    tprev[gid]  = make_float2(0.0f, 0.0f);
}

// ---------------- host-side launcher ----------------
extern "C" void kernel_launch(void* const* d_in, const int* in_sizes, int n_in,
                              void* d_out, int out_size, void* d_ws, size_t ws_size,
                              hipStream_t stream) {
    const float* x  = (const float*)d_in[0];  // (16,1,131072)
    const float* W1 = (const float*)d_in[1];  // (2,256)
    const float* b1 = (const float*)d_in[2];  // (256)
    const float* W2 = (const float*)d_in[3];  // (256,256)
    const float* b2 = (const float*)d_in[4];  // (256)
    const float* W3 = (const float*)d_in[5];  // (256,64)
    const float* b3 = (const float*)d_in[6];  // (64)

    float* rec     = (float*)d_out;                    // (16,131072)
    float* content = (float*)d_out + (long)NB * TLEN;  // (16,513,64)

    // workspace carve-up
    char* base = (char*)d_ws;
    size_t off = 0;
    auto carve = [&](size_t bytes) -> char* {
        char* p = base + off;
        off += (bytes + 255) & ~(size_t)255;
        return p;
    };
    float*    feats  = (float*)   carve((size_t)NROWS * 2 * 4);
    _Float16* W2p    = (_Float16*)carve((size_t)256 * 256 * 2);
    _Float16* W3p    = (_Float16*)carve((size_t)256 * 64 * 2);
    _Float16* h1h    = (_Float16*)carve((size_t)NROWS * 256 * 2);
    _Float16* h2h    = (_Float16*)carve((size_t)NROWS * 256 * 2);
    float*    fb     = (float*)   carve((size_t)NSTFT * NMELS * 4);
    float*    mag    = (float*)   carve((size_t)NROWS * NSTFT * 4);   // later reused as `lin`
    float*    mel    = (float*)   carve((size_t)NROWS * NMELS * 4);
    float2*   angles = (float2*)  carve((size_t)NROWS * NSTFT * 8);
    float2*   tprev  = (float2*)  carve((size_t)NROWS * NSTFT * 8);
    float*    frames = (float*)   carve((size_t)NROWS * NFFT * 4);
    float*    inv    = (float*)   carve((size_t)NB * TLEN * 4);
    float*    env    = (float*)   carve((size_t)LOLA * 4);
    float*    lin    = mag;  // alias: mag is dead after mel_kernel

    // --- features + MLP (WMMA) ---
    pitch_kernel<<<NROWS, 256, 0, stream>>>(x, feats);
    packB_kernel<<<(256 * 256 + 255) / 256, 256, 0, stream>>>(W2, W2p, 256, 256);
    packB_kernel<<<(256 * 64 + 255) / 256, 256, 0, stream>>>(W3, W3p, 256, 64);
    h1_kernel<<<(NROWS * 256 + 255) / 256, 256, 0, stream>>>(feats, W1, b1, h1h);
    gemm2_kernel<<<NROWS / 16, 256, 0, stream>>>(h1h, W2p, b2, h2h);
    gemm3_kernel<<<NROWS / 16, 128, 0, stream>>>(h2h, W3p, b3, content);

    // --- spectrogram -> mel -> linear ---
    stft_kernel<<<NROWS, 256, 0, stream>>>(x, 0, mag, nullptr, nullptr);
    fb_kernel<<<(NSTFT * NMELS + 255) / 256, 256, 0, stream>>>(fb);
    mel_kernel<<<(NROWS * NMELS + 255) / 256, 256, 0, stream>>>(mag, fb, mel);
    m2l_kernel<<<(int)(((long)NROWS * NSTFT + 255) / 256), 256, 0, stream>>>(mel, lin);

    // --- Griffin-Lim ---
    env_kernel<<<(LOLA + 255) / 256, 256, 0, stream>>>(env);
    ainit_kernel<<<(int)(((long)NROWS * NSTFT + 255) / 256), 256, 0, stream>>>(angles, tprev);
    for (int it = 0; it < 4; ++it) {
        istft_kernel<<<NROWS, 256, 0, stream>>>(lin, angles, frames);
        ola_kernel<<<(int)(((long)NB * TLEN + 255) / 256), 256, 0, stream>>>(frames, env, inv);
        stft_kernel<<<NROWS, 256, 0, stream>>>(inv, 1, nullptr, angles, tprev);
    }
    istft_kernel<<<NROWS, 256, 0, stream>>>(lin, angles, frames);
    ola_kernel<<<(int)(((long)NB * TLEN + 255) / 256), 256, 0, stream>>>(frames, env, rec);

    (void)in_sizes; (void)n_in; (void)out_size; (void)ws_size;
}
